// CLNF_54769422959177
// MI455X (gfx1250) — matrix operand order repaired and checked by reading, not compiled, not run on hardware
//
#include <hip/hip_runtime.h>

// MI455X / gfx1250: wave32, WMMA 16x16x32 f16->f32, big LDS (320KB/WGP).
// Fused forward+VJP of the 4-layer vector field.
// - activations live in LDS as f16 row-major -> A-fragments are two aligned
//   ds_load_b128 per K-step, no cvt on the WMMA feed path
// - weight tile staged into LDS pre-swizzled in B-fragment order (32B/frag)
// - p-tile fetched with GLOBAL_LOAD_ASYNC_TO_LDS_B128 (ASYNCcnt) overlapped
//   with weight staging
// - elementwise (sin-embed/softplus/layernorm fwd+bwd) in f32 VALU with
//   wave32 shuffle reductions

typedef __attribute__((ext_vector_type(16))) _Float16 v16h;
typedef __attribute__((ext_vector_type(8)))  _Float16 v8h;
typedef __attribute__((ext_vector_type(4)))  _Float16 v4h;
typedef __attribute__((ext_vector_type(8)))  float    v8f;
typedef __attribute__((ext_vector_type(4)))  float    v4f;

#define HDIM 128
#define ROWS 64          // rows per block (4 waves x 16 rows)
#define NTHREADS 128
#define NLAYER 4
#define LN_EPS 1e-5f

// dynamic-LDS byte offsets
#define OFF_ACT   0           // _Float16 [ROWS][HDIM]          16384
#define OFF_WT    16384       // _Float16 [HDIM][HDIM] swizzled 32768
#define OFF_Z     49152       // _Float16 [NLAYER][ROWS][HDIM]  65536 (head doubles as p-stage)
#define OFF_STATS 114688      // float    [NLAYER][ROWS][2]      2048
#define OFF_SEMB  116736      // float    [NLAYER][HDIM]         2048
#define OFF_BIAS  118784      // float    [HDIM]                  512
#define SMEM_BYTES 119296

__device__ __forceinline__ float softplus_f(float z) {
    return fmaxf(z, 0.0f) + log1pf(__expf(-fabsf(z)));
}
__device__ __forceinline__ float sigmoid_f(float z) {
    return 1.0f / (1.0f + __expf(-z));
}
__device__ __forceinline__ float wave_sum32(float v) {
    #pragma unroll
    for (int m = 16; m >= 1; m >>= 1) v += __shfl_xor(v, m, 32);
    return v;
}

// ---------------------------------------------------------------------------
// Weight staging in this code's B-fragment order (16-bit 32x16 B, contiguous k, matches A):
// lane<16: N=lane, K=j ; lane>=16: N=lane-16, K=16+j.
//   dstHalf((k,n)) = ((kt*8 + nt)*32 + lane)*16 + j
// transposed=true : B[k][n] = W[n][k]  (forward, x @ W^T)  -- v4f reads over k
// transposed=false: B[k][n] = W[k][n]  (backward, g @ W)   -- b32 reads over n
// Both write one v4h (b64) per 4 consecutive k.
// ---------------------------------------------------------------------------
__device__ __forceinline__ void stage_weight(_Float16* wt, const float* __restrict__ W,
                                             bool transposed) {
    for (int q = threadIdx.x; q < (HDIM * HDIM) / 4; q += NTHREADS) {
        int n, k0;
        v4f src;
        if (transposed) {
            n  = q >> 5;
            k0 = (q & 31) * 4;
            src = *(const v4f*)(W + n * HDIM + k0);           // coalesced over k
        } else {
            n  = q & (HDIM - 1);
            k0 = (q >> 7) * 4;
            src[0] = W[(k0 + 0) * HDIM + n];                  // coalesced over n
            src[1] = W[(k0 + 1) * HDIM + n];
            src[2] = W[(k0 + 2) * HDIM + n];
            src[3] = W[(k0 + 3) * HDIM + n];
        }
        const int kt   = k0 >> 5;
        const int kr   = k0 & 31;
        const int lane = ((kr >> 4) << 4) | (n & 15);
        const int dst  = (((kt << 3) | (n >> 4)) * 32 + lane) * 16 + (kr & 15);
        v4h d;
        d[0] = (_Float16)src[0]; d[1] = (_Float16)src[1];
        d[2] = (_Float16)src[2]; d[3] = (_Float16)src[3];
        *(v4h*)(wt + dst) = d;
    }
}

// ---------------------------------------------------------------------------
// Per-wave: [16 x 128] = [16 x 128](actIn,f16,LDS) x [128 x 128](wt,f16) + bias.
// A-fragment (16-bit A 16x32): lane holds row l16; K runs {hs*16+0..15}
// per 32-K step -> two aligned b128 loads.  C/D: VGPR c -> row c (+8 if hs),
// col = nt*16 + l16.
// ---------------------------------------------------------------------------
__device__ __forceinline__ void matmul16x128(const _Float16* actIn, const _Float16* wt,
        const float* bias, _Float16* outLds, float* outGlob, int gRowBase, float scale)
{
    const int lane = threadIdx.x & 31;
    const int wave = threadIdx.x >> 5;
    const int wr0  = wave * 16;
    const int hs   = lane >> 4;
    const int l16  = lane & 15;

    const _Float16* rowp = actIn + (wr0 + l16) * HDIM;
    v16h a[4];
    #pragma unroll
    for (int kt = 0; kt < 4; ++kt) {
        const _Float16* bp = rowp + kt * 32 + hs * 16;
        v8h r0 = *(const v8h*)(bp);
        v8h r1 = *(const v8h*)(bp + 8);
        a[kt] = __builtin_shufflevector(r0, r1, 0, 1, 2, 3, 4, 5, 6, 7,
                                                8, 9, 10, 11, 12, 13, 14, 15);
    }
    #pragma unroll
    for (int nt = 0; nt < 8; ++nt) {
        v8f acc = {};
        #pragma unroll
        for (int kt = 0; kt < 4; ++kt) {
            const v16h b = *(const v16h*)(wt + (((kt << 3) | nt) * 32 + lane) * 16);
            acc = __builtin_amdgcn_wmma_f32_16x16x32_f16(
                false, a[kt], false, b, (short)0, acc, false, false);
        }
        const int col = nt * 16 + l16;
        const float bv = bias[col];
        #pragma unroll
        for (int c = 0; c < 8; ++c) {
            const int row = wr0 + c + hs * 8;
            const float v = (acc[c] + bv) * scale;
            if (outGlob) outGlob[(size_t)(gRowBase + row) * HDIM + col] = v;
            else         outLds[row * HDIM + col] = (_Float16)v;
        }
    }
}

// forward elementwise: h += sin-embed; stash z; softplus; layernorm -> act (f16)
__device__ __forceinline__ void ew_forward(int layer, _Float16* act, _Float16* zbuf,
        float* stats, const float* semb, const float* __restrict__ gamma,
        const float* __restrict__ beta)
{
    const int lane = threadIdx.x & 31;
    const int wave = threadIdx.x >> 5;
    const int wr0  = wave * 16;
    const int c0   = lane * 4;

    const v4f sv = *(const v4f*)(semb + layer * HDIM + c0);
    const v4f gv = *(const v4f*)(gamma + layer * HDIM + c0);
    const v4f bv = *(const v4f*)(beta + layer * HDIM + c0);

    for (int r = 0; r < 16; ++r) {
        const int row = wr0 + r;
        const v4h xh16 = *(const v4h*)(act + row * HDIM + c0);
        v4h zz;
        float u[4];
        float s1 = 0.0f, s2 = 0.0f;
        #pragma unroll
        for (int j = 0; j < 4; ++j) {
            float z = (float)xh16[j] + sv[j];
            zz[j] = (_Float16)z;
            float uu = softplus_f(z);
            u[j] = uu;  s1 += uu;  s2 += uu * uu;
        }
        *(v4h*)(zbuf + (layer * ROWS + row) * HDIM + c0) = zz;
        s1 = wave_sum32(s1);
        s2 = wave_sum32(s2);
        const float mu   = s1 * (1.0f / HDIM);
        const float var  = s2 * (1.0f / HDIM) - mu * mu;
        const float rstd = rsqrtf(var + LN_EPS);
        if (lane == 0) {
            stats[(layer * ROWS + row) * 2 + 0] = mu;
            stats[(layer * ROWS + row) * 2 + 1] = rstd;
        }
        v4h ov;
        #pragma unroll
        for (int j = 0; j < 4; ++j)
            ov[j] = (_Float16)((u[j] - mu) * rstd * gv[j] + bv[j]);
        *(v4h*)(act + row * HDIM + c0) = ov;
    }
}

// backward elementwise through LN + softplus (sin-embed add: identity grad)
__device__ __forceinline__ void ew_backward(int layer, _Float16* act, const _Float16* zbuf,
        const float* stats, const float* __restrict__ gamma)
{
    const int lane = threadIdx.x & 31;
    const int wave = threadIdx.x >> 5;
    const int wr0  = wave * 16;
    const int c0   = lane * 4;

    const v4f gv = *(const v4f*)(gamma + layer * HDIM + c0);

    for (int r = 0; r < 16; ++r) {
        const int row  = wr0 + r;
        const float mu   = stats[(layer * ROWS + row) * 2 + 0];
        const float rstd = stats[(layer * ROWS + row) * 2 + 1];
        const v4h zz = *(const v4h*)(zbuf + (layer * ROWS + row) * HDIM + c0);
        const v4h gh = *(const v4h*)(act + row * HDIM + c0);
        float gx[4], xh[4], sg[4];
        float s1 = 0.0f, s2 = 0.0f;
        #pragma unroll
        for (int j = 0; j < 4; ++j) {
            float z  = (float)zz[j];
            float uu = softplus_f(z);
            xh[j] = (uu - mu) * rstd;
            sg[j] = sigmoid_f(z);
            gx[j] = (float)gh[j] * gv[j];
            s1 += gx[j];
            s2 += gx[j] * xh[j];
        }
        s1 = wave_sum32(s1) * (1.0f / HDIM);
        s2 = wave_sum32(s2) * (1.0f / HDIM);
        v4h ov;
        #pragma unroll
        for (int j = 0; j < 4; ++j)
            ov[j] = (_Float16)(rstd * (gx[j] - s1 - xh[j] * s2) * sg[j]);
        *(v4h*)(act + row * HDIM + c0) = ov;
    }
}

__global__ void __launch_bounds__(NTHREADS)
CLNF_54769422959177_kernel(
    const float* __restrict__ t,     const float* __restrict__ p,
    const float* __restrict__ w,     const float* __restrict__ W_in,
    const float* __restrict__ b_in,  const float* __restrict__ fw,
    const float* __restrict__ fb,    const float* __restrict__ gamma,
    const float* __restrict__ beta,  const float* __restrict__ Wl,
    const float* __restrict__ bl,    const float* __restrict__ W_out,
    const float* __restrict__ b_out, float* __restrict__ dp,
    float* __restrict__ dw)
{
    extern __shared__ char smem[];
    _Float16* act   = (_Float16*)(smem + OFF_ACT);
    _Float16* wt    = (_Float16*)(smem + OFF_WT);
    _Float16* zbuf  = (_Float16*)(smem + OFF_Z);
    float*    pstage= (float*)(smem + OFF_Z);      // zbuf is dead at kernel start
    float*    stats = (float*)(smem + OFF_STATS);
    float*    semb  = (float*)(smem + OFF_SEMB);
    float*    bias  = (float*)(smem + OFF_BIAS);

    const int tid  = threadIdx.x;
    const int row0 = blockIdx.x * ROWS;

    // async-copy the p tile into LDS staging (GLOBAL_LOAD_ASYNC_TO_LDS_B128,
    // tracked by ASYNCcnt) and overlap with weight/embedding staging.
    for (int i = tid; i < (ROWS * HDIM) / 4; i += NTHREADS) {
        unsigned ldsOff = (unsigned)(size_t)(pstage + i * 4);
        unsigned long long ga =
            (unsigned long long)(size_t)(p + (size_t)row0 * HDIM + i * 4);
        asm volatile("global_load_async_to_lds_b128 %0, %1, off"
                     :: "v"(ldsOff), "v"(ga) : "memory");
    }

    // warm L2 for the cotangent tile (global_prefetch_b8)
    __builtin_prefetch(w + (size_t)row0 * HDIM + tid * 64, 0, 1);

    const float tt = t[0];
    for (int e = tid; e < NLAYER * HDIM; e += NTHREADS)
        semb[e] = sinf(tt * fw[e] + fb[e]);
    stage_weight(wt, W_in, true);
    if (tid < HDIM) bias[tid] = b_in[tid];

    asm volatile("s_wait_asynccnt 0x0" ::: "memory");
    __syncthreads();

    // convert staged p (f32) -> act (f16)
    for (int i = tid; i < (ROWS * HDIM) / 4; i += NTHREADS) {
        v4f x = *(const v4f*)(pstage + i * 4);
        v4h d;
        d[0] = (_Float16)x[0]; d[1] = (_Float16)x[1];
        d[2] = (_Float16)x[2]; d[3] = (_Float16)x[3];
        *(v4h*)(act + i * 4) = d;
    }
    __syncthreads();

    // ---------------- forward ----------------
    matmul16x128(act, wt, bias, act, nullptr, 0, 1.0f);        // h0

    for (int i = 0; i < NLAYER; ++i) {
        if (i + 1 < NLAYER)
            __builtin_prefetch(Wl + (size_t)(i + 1) * HDIM * HDIM + tid * 128, 0, 1);
        ew_forward(i, act, zbuf, stats, semb, gamma, beta);    // wave-local
        __syncthreads();
        stage_weight(wt, Wl + (size_t)i * HDIM * HDIM, true);
        if (tid < HDIM) bias[tid] = bl[i * HDIM + tid];
        __syncthreads();
        matmul16x128(act, wt, bias, act, nullptr, 0, 1.0f);    // h_{i+1}
    }

    __syncthreads();
    stage_weight(wt, W_out, true);
    if (tid < HDIM) bias[tid] = b_out[tid];
    __syncthreads();
    matmul16x128(act, wt, bias, nullptr, dp, row0, 1.0f);      // dp = vf(p)

    // ---------------- backward (cotangent = w) ----------------
    __syncthreads();
    for (int i = tid; i < (ROWS * HDIM) / 4; i += NTHREADS) {  // w tile -> act f16
        v4f x = *(const v4f*)(w + (size_t)row0 * HDIM + i * 4);
        v4h d;
        d[0] = (_Float16)x[0]; d[1] = (_Float16)x[1];
        d[2] = (_Float16)x[2]; d[3] = (_Float16)x[3];
        *(v4h*)(act + i * 4) = d;
    }
    stage_weight(wt, W_out, false);                            // g_h4 = w @ W_out
    if (tid < HDIM) bias[tid] = 0.0f;                          // no bias in backward
    __syncthreads();
    matmul16x128(act, wt, bias, act, nullptr, 0, 1.0f);

    for (int i = NLAYER - 1; i >= 0; --i) {
        __syncthreads();
        stage_weight(wt, Wl + (size_t)i * HDIM * HDIM, false); // g_v = g @ Wl_i
        __syncthreads();
        matmul16x128(act, wt, bias, act, nullptr, 0, 1.0f);
        ew_backward(i, act, zbuf, stats, gamma);               // wave-local
    }

    __syncthreads();
    stage_weight(wt, W_in, false);                             // g_p = g @ W_in
    __syncthreads();
    matmul16x128(act, wt, bias, nullptr, dw, row0, -1.0f);     // dw = -g_p
}

extern "C" void kernel_launch(void* const* d_in, const int* in_sizes, int n_in,
                              void* d_out, int out_size, void* d_ws, size_t ws_size,
                              hipStream_t stream) {
    (void)n_in; (void)out_size; (void)d_ws; (void)ws_size;
    const float* t     = (const float*)d_in[0];
    const float* p     = (const float*)d_in[1];
    const float* w     = (const float*)d_in[2];
    const float* W_in  = (const float*)d_in[3];
    const float* b_in  = (const float*)d_in[4];
    const float* fw    = (const float*)d_in[5];
    const float* fb    = (const float*)d_in[6];
    const float* gamma = (const float*)d_in[7];
    const float* beta  = (const float*)d_in[8];
    const float* Wl    = (const float*)d_in[9];
    const float* bl    = (const float*)d_in[10];
    const float* W_out = (const float*)d_in[11];
    const float* b_out = (const float*)d_in[12];

    float* out = (float*)d_out;
    const int Btot = in_sizes[1] / HDIM;          // 131072
    float* dp = out;
    float* dw = out + (size_t)Btot * HDIM;

    hipFuncSetAttribute((const void*)CLNF_54769422959177_kernel,
                        hipFuncAttributeMaxDynamicSharedMemorySize, SMEM_BYTES);

    const int nblk = (Btot + ROWS - 1) / ROWS;    // 2048 blocks
    CLNF_54769422959177_kernel<<<nblk, NTHREADS, SMEM_BYTES, stream>>>(
        t, p, w, W_in, b_in, fw, fb, gamma, beta, Wl, bl, W_out, b_out, dp, dw);
}
